// GAT_54417235640670
// MI455X (gfx1250) — matrix operand order repaired and checked by reading, not compiled, not run on hardware
//
#include <hip/hip_runtime.h>
#include <hip/hip_bf16.h>
#include <cstdint>

#define NEG_SLOPE 0.2f

typedef __attribute__((ext_vector_type(16))) __bf16 v16bf;
typedef __attribute__((ext_vector_type(8)))  float  v8f;

// ---------------------------------------------------------------------------
// fp32 -> bf16 cast (RNE via clang __bf16 conversion)
// ---------------------------------------------------------------------------
__global__ __launch_bounds__(256) void cast_f32_bf16(const float* __restrict__ in,
                                                     __bf16* __restrict__ out, int n) {
  int i = blockIdx.x * blockDim.x + threadIdx.x;
  if (i < n) out[i] = (__bf16)in[i];
}

// ---------------------------------------------------------------------------
// W[K][Nc] fp32  ->  Wt[Nc][K] bf16 (transposed so B fragments are contiguous)
// ---------------------------------------------------------------------------
__global__ __launch_bounds__(256) void cast_transpose_w(const float* __restrict__ W,
                                                        __bf16* __restrict__ Wt,
                                                        int K, int Nc) {
  int i = blockIdx.x * blockDim.x + threadIdx.x;
  if (i >= K * Nc) return;
  int k = i / Nc;
  int n = i - k * Nc;
  Wt[(size_t)n * K + k] = (__bf16)W[i];
}

// ---------------------------------------------------------------------------
// WMMA GEMM: C[M,Nc] = A[M,K] (bf16, row-major) * B[K,Nc] given as
// Wt[Nc,K] (bf16, i.e. B transposed). fp32 accumulate.
// One wave computes a 16x64 strip: 4 accumulators reuse each A fragment.
// K compile-time (128 here) -> fully unrolled, 16 v_wmma per wave.
// Fragment packing per CDNA5 ISA 7.12.2:
//   A lane half 0: k = kc+{0..7, 16..23}; half 1: kc+{8..15, 24..31}
//   B lane = column; half 0: k = kc+{0..7, 16..23}; half 1: kc+{8..15, 24..31}
// All fragment loads are aligned 16B -> global_load_b128.
// ---------------------------------------------------------------------------
template <int K>
__global__ __launch_bounds__(32) void wmma_gemm_bf16(const __bf16* __restrict__ A,
                                                     const __bf16* __restrict__ Wt,
                                                     float* __restrict__ C,
                                                     int M, int Nc) {
  const int tileM = blockIdx.x;
  const int nBase = blockIdx.y * 64;
  const int lane  = threadIdx.x;
  const int half  = lane >> 4;   // 0: lanes 0-15, 1: lanes 16-31
  const int l16   = lane & 15;

  const int row = tileM * 16 + l16;
  const uint4* __restrict__ arow = (const uint4*)(A + (size_t)row * K);

  v8f acc[4] = {};

#pragma unroll
  for (int kc = 0; kc < K; kc += 32) {
    union { v16bf v; uint4 q[2]; } af;
    af.q[0] = arow[(kc + half * 8) >> 3];        // 8 bf16 = 16B
    af.q[1] = arow[(kc + half * 8 + 16) >> 3];
#pragma unroll
    for (int t = 0; t < 4; ++t) {
      const int n = nBase + t * 16 + l16;
      const uint4* __restrict__ bcol =
          (const uint4*)(Wt + (size_t)n * K + kc + half * 8);
      union { v16bf v; uint4 q[2]; } bfr;
      bfr.q[0] = bcol[0];
      bfr.q[1] = bcol[2];
      acc[t] = __builtin_amdgcn_wmma_f32_16x16x32_bf16(
          /*neg_a=*/false, af.v, /*neg_b=*/false, bfr.v,
          /*c_mod=*/(short)0, acc[t], /*reuse_a=*/false, /*reuse_b=*/false);
    }
  }

  // C layout: VGPR r -> M = r + half*8, N = l16
#pragma unroll
  for (int t = 0; t < 4; ++t) {
#pragma unroll
    for (int r = 0; r < 8; ++r) {
      const int m = tileM * 16 + r + half * 8;
      C[(size_t)m * Nc + nBase + t * 16 + l16] = acc[t][r];
    }
  }
}

// ---------------------------------------------------------------------------
// Per-node attention scores: s_src[i] = h[i].a_src, s_dst[i] = h[i].a_dst
// One wave per node (4 waves / 128-thread block).
// ---------------------------------------------------------------------------
__global__ __launch_bounds__(128) void node_scores(const float* __restrict__ H,
                                                   const float* __restrict__ a_s,
                                                   const float* __restrict__ a_d,
                                                   float* __restrict__ s_src,
                                                   float* __restrict__ s_dst,
                                                   int N, int F) {
  int wave = threadIdx.x >> 5;
  int lane = threadIdx.x & 31;
  int node = blockIdx.x * 4 + wave;
  if (node >= N) return;
  float ss = 0.f, sd = 0.f;
  for (int f = lane; f < F; f += 32) {
    float h = H[(size_t)node * F + f];
    ss += h * a_s[f];
    sd += h * a_d[f];
  }
#pragma unroll
  for (int off = 16; off; off >>= 1) {
    ss += __shfl_down(ss, off, 32);
    sd += __shfl_down(sd, off, 32);
  }
  if (lane == 0) { s_src[node] = ss; s_dst[node] = sd; }
}

// ---------------------------------------------------------------------------
// Init kernels
// ---------------------------------------------------------------------------
__global__ __launch_bounds__(256) void init_softmax_state(float* __restrict__ mmax,
                                                          float* __restrict__ denom, int N) {
  int i = blockIdx.x * blockDim.x + threadIdx.x;
  if (i < N) { mmax[i] = -INFINITY; denom[i] = 0.f; }
}

__global__ __launch_bounds__(256) void zero_f32(float* __restrict__ p, int n) {
  int i = blockIdx.x * blockDim.x + threadIdx.x;
  if (i < n) p[i] = 0.f;
}

// Monotonic float atomic max via int/uint trick (valid with -inf init).
__device__ inline void atomicMaxF32(float* addr, float val) {
  if (val >= 0.f) atomicMax((int*)addr, __float_as_int(val));
  else            atomicMin((unsigned int*)addr, __float_as_uint(val));
}

// ---------------------------------------------------------------------------
// Edge pass 1: leaky-relu logit + segment max (by dst)
// Edges e < E come from edge_index; e >= E are self-loops (node e-E).
// ---------------------------------------------------------------------------
__global__ __launch_bounds__(256) void edge_logits(const int* __restrict__ ei,
                                                   const float* __restrict__ s_src,
                                                   const float* __restrict__ s_dst,
                                                   float* __restrict__ logit,
                                                   float* __restrict__ mmax,
                                                   int E, int Etot) {
  int e = blockIdx.x * blockDim.x + threadIdx.x;
  if (e >= Etot) return;
  int s = (e < E) ? ei[e]     : (e - E);
  int d = (e < E) ? ei[E + e] : (e - E);
  float x = s_src[s] + s_dst[d];
  x = (x > 0.f) ? x : NEG_SLOPE * x;
  logit[e] = x;
  atomicMaxF32(&mmax[d], x);
}

// ---------------------------------------------------------------------------
// Edge pass 2: e = exp(logit - m[dst]); denom[dst] += e  (logit overwritten)
// ---------------------------------------------------------------------------
__global__ __launch_bounds__(256) void edge_expsum(const int* __restrict__ ei,
                                                   float* __restrict__ logit,
                                                   const float* __restrict__ mmax,
                                                   float* __restrict__ denom,
                                                   int E, int Etot) {
  int e = blockIdx.x * blockDim.x + threadIdx.x;
  if (e >= Etot) return;
  int d = (e < E) ? ei[E + e] : (e - E);
  float v = __expf(logit[e] - mmax[d]);
  logit[e] = v;
  atomicAdd(&denom[d], v);
}

// ---------------------------------------------------------------------------
// Edge pass 3: agg[dst,:] += (e/denom[dst]) * H[src,:]
// One 128-thread block per edge: coalesced row gather + per-feature atomics.
// ---------------------------------------------------------------------------
__global__ __launch_bounds__(128) void edge_scatter(const int* __restrict__ ei,
                                                    const float* __restrict__ logit,
                                                    const float* __restrict__ denom,
                                                    const float* __restrict__ H,
                                                    float* __restrict__ agg,
                                                    int E, int Etot, int F) {
  int e = blockIdx.x;
  if (e >= Etot) return;
  int s = (e < E) ? ei[e]     : (e - E);
  int d = (e < E) ? ei[E + e] : (e - E);
  float alpha = logit[e] / denom[d];
  const float* hrow = H + (size_t)s * F;
  float* arow = agg + (size_t)d * F;
  for (int f = threadIdx.x; f < F; f += blockDim.x)
    atomicAdd(&arow[f], alpha * hrow[f]);
}

// ---------------------------------------------------------------------------
// agg = relu(agg + bias), in place (fp32 activation for next layer / output)
// ---------------------------------------------------------------------------
__global__ __launch_bounds__(256) void bias_relu(float* __restrict__ agg,
                                                 const float* __restrict__ b,
                                                 int total, int F) {
  int i = blockIdx.x * blockDim.x + threadIdx.x;
  if (i >= total) return;
  float v = agg[i] + b[i & (F - 1)];   // F is a power of two (128 / 64)
  agg[i] = (v > 0.f) ? v : 0.f;
}

// ---------------------------------------------------------------------------
// log_softmax over 64 classes: one wave per node, 2 classes per lane.
// ---------------------------------------------------------------------------
__global__ __launch_bounds__(128) void log_softmax64(const float* __restrict__ in,
                                                     float* __restrict__ out, int N) {
  int wave = threadIdx.x >> 5;
  int lane = threadIdx.x & 31;
  int node = blockIdx.x * 4 + wave;
  if (node >= N) return;
  const float* row = in + (size_t)node * 64;
  float a = row[lane];
  float b = row[32 + lane];
  float mx = fmaxf(a, b);
#pragma unroll
  for (int off = 16; off; off >>= 1) mx = fmaxf(mx, __shfl_xor(mx, off, 32));
  float se = __expf(a - mx) + __expf(b - mx);
#pragma unroll
  for (int off = 16; off; off >>= 1) se += __shfl_xor(se, off, 32);
  float lse = mx + __logf(se);
  float* orow = out + (size_t)node * 64;
  orow[lane]      = a - lse;
  orow[32 + lane] = b - lse;
}

// ---------------------------------------------------------------------------
// Host orchestration
// ---------------------------------------------------------------------------
static inline int cdiv(int a, int b) { return (a + b - 1) / b; }

struct GatBufs {
  __bf16* abf;   // [N, 128] bf16 activation
  __bf16* wtbf;  // [128, 128] bf16 weights, TRANSPOSED (Wt[Nc][K])
  float*  H;     // [N, 128] transformed features
  float*  s_src; // [N]
  float*  s_dst; // [N]
  float*  mmax;  // [N]
  float*  denom; // [N]
  float*  elog;  // [E + N]
  float*  agg;   // [N, 128]
};

static void gat_layer(const float* act, const float* W, const float* a_s,
                      const float* a_d, const float* bias, const int* ei,
                      int N, int E, int Fin, int Fout,
                      const GatBufs& ws, hipStream_t stream) {
  const int Etot = E + N;

  cast_f32_bf16<<<cdiv(N * Fin, 256), 256, 0, stream>>>(act, ws.abf, N * Fin);
  cast_transpose_w<<<cdiv(Fin * Fout, 256), 256, 0, stream>>>(W, ws.wtbf, Fin, Fout);

  // K = Fin is always 128 for this network (F_IN = H = 128).
  dim3 ggrid(N / 16, Fout / 64);
  wmma_gemm_bf16<128><<<ggrid, 32, 0, stream>>>(ws.abf, ws.wtbf, ws.H, N, Fout);

  node_scores<<<cdiv(N, 4), 128, 0, stream>>>(ws.H, a_s, a_d, ws.s_src, ws.s_dst, N, Fout);
  init_softmax_state<<<cdiv(N, 256), 256, 0, stream>>>(ws.mmax, ws.denom, N);
  zero_f32<<<cdiv(N * Fout, 256), 256, 0, stream>>>(ws.agg, N * Fout);

  edge_logits<<<cdiv(Etot, 256), 256, 0, stream>>>(ei, ws.s_src, ws.s_dst,
                                                   ws.elog, ws.mmax, E, Etot);
  edge_expsum<<<cdiv(Etot, 256), 256, 0, stream>>>(ei, ws.elog, ws.mmax, ws.denom, E, Etot);
  edge_scatter<<<Etot, 128, 0, stream>>>(ei, ws.elog, ws.denom, ws.H, ws.agg, E, Etot, Fout);

  bias_relu<<<cdiv(N * Fout, 256), 256, 0, stream>>>(ws.agg, bias, N * Fout, Fout);
}

extern "C" void kernel_launch(void* const* d_in, const int* in_sizes, int n_in,
                              void* d_out, int out_size, void* d_ws, size_t ws_size,
                              hipStream_t stream) {
  // setup_inputs order:
  // 0:x 1:edge_index 2:W1 3:as1 4:ad1 5:b1 6:W2 7:as2 8:ad2 9:b2 10:W3 11:as3 12:ad3 13:b3
  const float* x  = (const float*)d_in[0];
  const int*   ei = (const int*)d_in[1];
  const float* W1 = (const float*)d_in[2];
  const float* as1 = (const float*)d_in[3];
  const float* ad1 = (const float*)d_in[4];
  const float* b1 = (const float*)d_in[5];
  const float* W2 = (const float*)d_in[6];
  const float* as2 = (const float*)d_in[7];
  const float* ad2 = (const float*)d_in[8];
  const float* b2 = (const float*)d_in[9];
  const float* W3 = (const float*)d_in[10];
  const float* as3 = (const float*)d_in[11];
  const float* ad3 = (const float*)d_in[12];
  const float* b3 = (const float*)d_in[13];

  const int H1  = in_sizes[3];            // 128
  const int Fin = in_sizes[2] / H1;       // 128
  const int N   = in_sizes[0] / Fin;      // 50000
  const int E   = in_sizes[1] / 2;        // 1600000
  const int C   = in_sizes[11];           // 64
  const int Fmax = (Fin > H1) ? Fin : H1; // 128

  // Workspace carve-out (256B aligned slices)
  char* w = (char*)d_ws;
  auto carve = [&](size_t bytes) -> void* {
    void* p = (void*)w;
    w += (bytes + 255) & ~(size_t)255;
    return p;
  };
  GatBufs ws;
  ws.abf   = (__bf16*)carve((size_t)N * Fmax * sizeof(__bf16));
  ws.wtbf  = (__bf16*)carve((size_t)Fmax * Fmax * sizeof(__bf16));
  ws.H     = (float*)carve((size_t)N * Fmax * sizeof(float));
  ws.s_src = (float*)carve((size_t)N * sizeof(float));
  ws.s_dst = (float*)carve((size_t)N * sizeof(float));
  ws.mmax  = (float*)carve((size_t)N * sizeof(float));
  ws.denom = (float*)carve((size_t)N * sizeof(float));
  ws.elog  = (float*)carve((size_t)(E + N) * sizeof(float));
  ws.agg   = (float*)carve((size_t)N * Fmax * sizeof(float));

  // Layer 1: x -> agg (relu'd)
  gat_layer(x, W1, as1, ad1, b1, ei, N, E, Fin, H1, ws, stream);
  // Layer 2: agg -> agg (abf is re-cast from agg before agg is rezeroed)
  gat_layer(ws.agg, W2, as2, ad2, b2, ei, N, E, H1, H1, ws, stream);
  // Layer 3: agg -> agg, Fout = C
  gat_layer(ws.agg, W3, as3, ad3, b3, ei, N, E, H1, C, ws, stream);

  // log_softmax over C=64 classes -> d_out
  log_softmax64<<<cdiv(N, 4), 128, 0, stream>>>(ws.agg, (float*)d_out, N);
}
